// CrossAttentionDoggettx_63823214019168
// MI455X (gfx1250) — matrix operand order repaired and checked by reading, not compiled, not run on hardware
//
#include <hip/hip_runtime.h>
#include <hip/hip_bf16.h>

typedef _Float16 v16h __attribute__((ext_vector_type(16)));
typedef _Float16 v8h  __attribute__((ext_vector_type(8)));
typedef float    v8f  __attribute__((ext_vector_type(8)));

#define SEQ 2048
#define DM  1024
#define NH  16
#define DH  64
#define NBATCH 2
#define MTOT (NBATCH * SEQ)   // 4096
#define ATTN_SCALE 0.125f     // 64^-0.5

// ---------------------------------------------------------------------------
// WMMA helpers (CDNA5 v_wmma_f32_16x16x32_f16, wave32)
// ---------------------------------------------------------------------------
__device__ __forceinline__ v8f wmma_f16(v16h a, v16h b, v8f c) {
  return __builtin_amdgcn_wmma_f32_16x16x32_f16(
      /*neg_a=*/false, a, /*neg_b=*/false, b,
      /*c_mod=*/(short)0, c, /*reuse_a=*/false, /*reuse_b=*/false);
}

// A-matrix 16x32 f16 fragment from row-major memory (row stride ld halves).
// lane: row = lane&15 ; K = 16*(lane>=16)+j (contiguous)
__device__ __forceinline__ v16h load_a16x32(const _Float16* __restrict__ p, int ld) {
  const int lane = threadIdx.x & 31;
  const int row  = lane & 15;
  const int ko   = (lane >> 4) << 4;   // 0 or 16
  const _Float16* b0 = p + (size_t)row * ld + ko;
  v8h lo = *(const v8h*)(b0);
  v8h hi = *(const v8h*)(b0 + 8);
  v16h r;
#pragma unroll
  for (int t = 0; t < 8; ++t) { r[t] = lo[t]; r[t + 8] = hi[t]; }
  return r;
}

// B-matrix 32x16 f16 fragment; memory holds B^T row-major: column `c` of B is
// the contiguous row `c` at p + c*ld. lane: col = lane&15 ; K = 16*(lane>=16)+j
__device__ __forceinline__ v16h load_b32x16(const _Float16* __restrict__ p, int ld) {
  const int lane = threadIdx.x & 31;
  const int col  = lane & 15;
  const int ko   = (lane >> 4) << 4;   // 0 or 16
  return *(const v16h*)(p + (size_t)col * ld + ko);
}

// ---------------------------------------------------------------------------
// Stage 1: f32 -> f16 conversion
// ---------------------------------------------------------------------------
__global__ void cvt_f32_f16(const float* __restrict__ s, _Float16* __restrict__ d, int n) {
  int i = blockIdx.x * blockDim.x + threadIdx.x;
  int stride = gridDim.x * blockDim.x;
  for (; i < n; i += stride) d[i] = (_Float16)s[i];
}

// ---------------------------------------------------------------------------
// Stage 2: fused QKV projection. y = x @ W^T ; epilogue scatters into
// per-head layouts. blockIdx.z selects W in {Wq,Wk,Wv}.
// Wave tile 64x64 (4x4 WMMAs, 1 b128-load per WMMA).
// Grid: (DM/128, MTOT/256, 3), block 256 = 8 waves arranged 4(m) x 2(n).
// ---------------------------------------------------------------------------
__global__ __launch_bounds__(256)
void proj_qkv_kernel(const _Float16* __restrict__ xh,
                     const _Float16* __restrict__ wqh,
                     const _Float16* __restrict__ wkh,
                     const _Float16* __restrict__ wvh,
                     _Float16* __restrict__ qd,   // [B*H][SEQ][DH], pre-scaled
                     _Float16* __restrict__ kd,   // [B*H][SEQ][DH]
                     _Float16* __restrict__ vtd)  // [B*H][DH][SEQ] (transposed)
{
  const int which = blockIdx.z;
  const _Float16* wh = (which == 0) ? wqh : ((which == 1) ? wkh : wvh);
  const int wave = threadIdx.x >> 5;
  const int m0 = blockIdx.y * 256 + (wave & 3) * 64;
  const int n0 = blockIdx.x * 128 + (wave >> 2) * 64;

  v8f acc[4][4] = {};
  for (int kk = 0; kk < DM; kk += 32) {
    v16h a[4], b[4];
#pragma unroll
    for (int i = 0; i < 4; ++i)
      a[i] = load_a16x32(xh + (size_t)(m0 + 16 * i) * DM + kk, DM);
#pragma unroll
    for (int j = 0; j < 4; ++j)
      b[j] = load_b32x16(wh + (size_t)(n0 + 16 * j) * DM + kk, DM);
#pragma unroll
    for (int i = 0; i < 4; ++i)
#pragma unroll
      for (int j = 0; j < 4; ++j)
        acc[i][j] = wmma_f16(a[i], b[j], acc[i][j]);
  }

  const int lane   = threadIdx.x & 31;
  const int colL   = lane & 15;
  const int rowoff = (lane >> 4) << 3;
#pragma unroll
  for (int ti = 0; ti < 4; ++ti) {
#pragma unroll
    for (int tj = 0; tj < 4; ++tj) {
      const int col = n0 + tj * 16 + colL;   // = head*64 + d
      const int hh = col >> 6, dd = col & 63;
#pragma unroll
      for (int r = 0; r < 8; ++r) {
        const int row = m0 + ti * 16 + rowoff + r;   // = b*SEQ + n
        const int bb = row >> 11, nn = row & (SEQ - 1);
        const int bh = bb * NH + hh;
        const float v = acc[ti][tj][r];
        if (which == 0)
          qd[((size_t)bh * SEQ + nn) * DH + dd] = (_Float16)(v * ATTN_SCALE);
        else if (which == 1)
          kd[((size_t)bh * SEQ + nn) * DH + dd] = (_Float16)v;
        else
          vtd[((size_t)bh * DH + dd) * SEQ + nn] = (_Float16)v;
      }
    }
  }
}

// ---------------------------------------------------------------------------
// Stage 3: flash attention. One wave handles 32 query rows (2 tiles) of one
// (b,h): K-tile and V^T-tile fragments are shared across both query tiles.
// S^T = K_tile x Q^T (keys in VGPR rows -> cheap softmax reduction),
// online softmax, P via per-wave LDS relayout, O^T += V^T x P^T.
// Grid: (B*H, SEQ/128), block 128 (4 independent waves).
// ---------------------------------------------------------------------------
__global__ __launch_bounds__(128)
void attn_kernel(const _Float16* __restrict__ qd,
                 const _Float16* __restrict__ kd,
                 const _Float16* __restrict__ vtd,
                 _Float16* __restrict__ od)      // [B*SEQ][DM] f16
{
  __shared__ __align__(32) _Float16 plds[4][2][16 * 32];  // [wave][qt] P[i][j]

  const int wave = threadIdx.x >> 5;
  const int lane = threadIdx.x & 31;
  const int bh = blockIdx.x;                       // b*NH + h
  const int i0 = (blockIdx.y * 4 + wave) * 32;     // query row base (32 rows)

  const _Float16* Q  = qd  + (size_t)bh * SEQ * DH;
  const _Float16* Kp = kd  + (size_t)bh * SEQ * DH;
  const _Float16* VT = vtd + (size_t)bh * DH * SEQ;

  // Q^T as B-operand: 2 query tiles x 2 dh chunks, pinned in registers
  v16h qb[2][2];
#pragma unroll
  for (int qt = 0; qt < 2; ++qt) {
    qb[qt][0] = load_b32x16(Q + (size_t)(i0 + 16 * qt) * DH + 0,  DH);
    qb[qt][1] = load_b32x16(Q + (size_t)(i0 + 16 * qt) * DH + 32, DH);
  }

  const int iL    = lane & 15;          // query column this lane owns
  const int jbase = (lane >> 4) << 3;   // key sub-offset in D layout
  const int kbase = (lane >> 4) << 4;   // K offset in B layout

  float m[2] = { -1e30f, -1e30f };
  float l[2] = { 0.f, 0.f };
  v8f acc[2][4] = {};                   // [qt][d-block of 16] O^T accumulators

  for (int jb = 0; jb < SEQ; jb += 32) {
    // prefetch next key/value tiles into cache while we compute
    if (jb + 32 < SEQ) {
      __builtin_prefetch(Kp + (size_t)(jb + 32) * DH, 0, 3);
      __builtin_prefetch(VT + (size_t)0 * SEQ + jb + 32, 0, 3);
    }
    // ---- S^T tiles: [qt][key subtile], each 16 keys x 16 queries ----
    v8f st[2][2];
#pragma unroll
    for (int t = 0; t < 2; ++t) {
      const int j0 = jb + t * 16;
      v16h ka0 = load_a16x32(Kp + (size_t)j0 * DH + 0,  DH);
      v16h ka1 = load_a16x32(Kp + (size_t)j0 * DH + 32, DH);
#pragma unroll
      for (int qt = 0; qt < 2; ++qt) {
        v8f s = {};
        s = wmma_f16(ka0, qb[qt][0], s);
        s = wmma_f16(ka1, qb[qt][1], s);
        st[qt][t] = s;
      }
    }
    // ---- online softmax over keys (VGPR rows + one lane-half swap) ----
    float alpha[2];
    v16h pb[2];
#pragma unroll
    for (int qt = 0; qt < 2; ++qt) {
      float tm = -1e30f;
#pragma unroll
      for (int t = 0; t < 2; ++t)
#pragma unroll
        for (int r = 0; r < 8; ++r) tm = fmaxf(tm, st[qt][t][r]);
      tm = fmaxf(tm, __shfl_xor(tm, 16, 32));
      const float mnew = fmaxf(m[qt], tm);
      alpha[qt] = __expf(m[qt] - mnew);
      float rs = 0.f;
      float pv[2][8];
#pragma unroll
      for (int t = 0; t < 2; ++t)
#pragma unroll
        for (int r = 0; r < 8; ++r) {
          const float p = __expf(st[qt][t][r] - mnew);
          pv[t][r] = p;
          rs += p;
        }
      rs += __shfl_xor(rs, 16, 32);
      l[qt] = l[qt] * alpha[qt] + rs;
      m[qt] = mnew;
      // P -> LDS as f16 in [i][j] order (D layout -> B layout bridge)
#pragma unroll
      for (int t = 0; t < 2; ++t)
#pragma unroll
        for (int pr = 0; pr < 4; ++pr) {
          union { _Float16 hh[2]; unsigned u; } pk;
          pk.hh[0] = (_Float16)pv[t][2 * pr];
          pk.hh[1] = (_Float16)pv[t][2 * pr + 1];
          *(unsigned*)&plds[wave][qt][iL * 32 + t * 16 + jbase + 2 * pr] = pk.u;
        }
      // reload as B-fragment: B[k=j][col=i] = plds[i*32 + j]
      pb[qt] = *(const v16h*)&plds[wave][qt][iL * 32 + kbase];
    }
    // ---- O^T = O^T*alpha + V^T x P^T ; V^T frag shared by both qt ----
#pragma unroll
    for (int dt = 0; dt < 4; ++dt) {
      v16h va = load_a16x32(VT + (size_t)(dt * 16) * SEQ + jb, SEQ);
#pragma unroll
      for (int qt = 0; qt < 2; ++qt) {
#pragma unroll
        for (int r = 0; r < 8; ++r) acc[qt][dt][r] *= alpha[qt];
        acc[qt][dt] = wmma_f16(va, pb[qt], acc[qt][dt]);
      }
    }
  }

  // ---- finalize: O[i][d] = O^T[d][i]/l, store f16 row-major [tok][DM] ----
  const int bb = bh >> 4, hh = bh & 15;
  const int rowoff = (lane >> 4) << 3;
#pragma unroll
  for (int qt = 0; qt < 2; ++qt) {
    const float inv = 1.f / l[qt];
    const size_t rowbase =
        ((size_t)bb * SEQ + i0 + 16 * qt + iL) * DM + (size_t)hh * DH;
#pragma unroll
    for (int dt = 0; dt < 4; ++dt) {
#pragma unroll
      for (int pr = 0; pr < 4; ++pr) {
        union { _Float16 h2[2]; unsigned u; } pk;
        pk.h2[0] = (_Float16)(acc[qt][dt][2 * pr] * inv);
        pk.h2[1] = (_Float16)(acc[qt][dt][2 * pr + 1] * inv);
        *(unsigned*)&od[rowbase + dt * 16 + rowoff + 2 * pr] = pk.u;
      }
    }
  }
}

// ---------------------------------------------------------------------------
// Stage 4: output projection  out = O @ Wout^T + bout  (f32 result)
// Wave tile 64x64. Grid: (DM/128, MTOT/256), block 256.
// ---------------------------------------------------------------------------
__global__ __launch_bounds__(256)
void out_proj_kernel(const _Float16* __restrict__ oh,
                     const _Float16* __restrict__ woh,
                     const float* __restrict__ bout,
                     float* __restrict__ out)
{
  const int wave = threadIdx.x >> 5;
  const int m0 = blockIdx.y * 256 + (wave & 3) * 64;
  const int n0 = blockIdx.x * 128 + (wave >> 2) * 64;

  v8f acc[4][4] = {};
  for (int kk = 0; kk < DM; kk += 32) {
    v16h a[4], b[4];
#pragma unroll
    for (int i = 0; i < 4; ++i)
      a[i] = load_a16x32(oh + (size_t)(m0 + 16 * i) * DM + kk, DM);
#pragma unroll
    for (int j = 0; j < 4; ++j)
      b[j] = load_b32x16(woh + (size_t)(n0 + 16 * j) * DM + kk, DM);
#pragma unroll
    for (int i = 0; i < 4; ++i)
#pragma unroll
      for (int j = 0; j < 4; ++j)
        acc[i][j] = wmma_f16(a[i], b[j], acc[i][j]);
  }

  const int lane   = threadIdx.x & 31;
  const int colL   = lane & 15;
  const int rowoff = (lane >> 4) << 3;
#pragma unroll
  for (int tj = 0; tj < 4; ++tj) {
    const int col = n0 + tj * 16 + colL;
    const float bias = bout[col];
#pragma unroll
    for (int ti = 0; ti < 4; ++ti) {
#pragma unroll
      for (int r = 0; r < 8; ++r) {
        const int row = m0 + ti * 16 + rowoff + r;
        out[(size_t)row * DM + col] = acc[ti][tj][r] + bias;
      }
    }
  }
}

// ---------------------------------------------------------------------------
// Launcher
// ---------------------------------------------------------------------------
extern "C" void kernel_launch(void* const* d_in, const int* in_sizes, int n_in,
                              void* d_out, int out_size, void* d_ws, size_t ws_size,
                              hipStream_t stream) {
  const float* x    = (const float*)d_in[0];
  const float* Wq   = (const float*)d_in[1];
  const float* Wk   = (const float*)d_in[2];
  const float* Wv   = (const float*)d_in[3];
  const float* Wout = (const float*)d_in[4];
  const float* bout = (const float*)d_in[5];
  float* out = (float*)d_out;

  _Float16* ws = (_Float16*)d_ws;
  size_t off = 0;
  _Float16* xh  = ws + off; off += (size_t)MTOT * DM;      // 4M halves
  _Float16* wqh = ws + off; off += (size_t)DM * DM;        // 1M
  _Float16* wkh = ws + off; off += (size_t)DM * DM;
  _Float16* wvh = ws + off; off += (size_t)DM * DM;
  _Float16* woh = ws + off; off += (size_t)DM * DM;
  _Float16* qd  = ws + off; off += (size_t)MTOT * DM;      // [B*H][SEQ][DH]
  _Float16* kd  = ws + off; off += (size_t)MTOT * DM;
  _Float16* vtd = ws + off; off += (size_t)MTOT * DM;      // [B*H][DH][SEQ]
  _Float16* oh  = ws + off; off += (size_t)MTOT * DM;      // [tok][DM]

  // Stage 1: precision conversion (L2-resident afterwards)
  cvt_f32_f16<<<1024, 256, 0, stream>>>(x, xh, MTOT * DM);
  cvt_f32_f16<<<256, 256, 0, stream>>>(Wq, wqh, DM * DM);
  cvt_f32_f16<<<256, 256, 0, stream>>>(Wk, wkh, DM * DM);
  cvt_f32_f16<<<256, 256, 0, stream>>>(Wv, wvh, DM * DM);
  cvt_f32_f16<<<256, 256, 0, stream>>>(Wout, woh, DM * DM);

  // Stage 2: fused QKV projection (64x64 wave tiles)
  proj_qkv_kernel<<<dim3(DM / 128, MTOT / 256, 3), 256, 0, stream>>>(
      xh, wqh, wkh, wvh, qd, kd, vtd);

  // Stage 3: flash attention (32 queries per wave)
  attn_kernel<<<dim3(NBATCH * NH, SEQ / 128), 128, 0, stream>>>(qd, kd, vtd, oh);

  // Stage 4: output projection + bias (64x64 wave tiles)
  out_proj_kernel<<<dim3(DM / 128, MTOT / 256), 256, 0, stream>>>(oh, woh, bout, out);
}
